// Net_33294586479042
// MI455X (gfx1250) — matrix operand order repaired and checked by reading, not compiled, not run on hardware
//
#include <hip/hip_runtime.h>
#include <stdint.h>

typedef __attribute__((ext_vector_type(16))) _Float16 v16h;
typedef __attribute__((ext_vector_type(8)))  float    v8f;

#define HDIM 512

// ---------------------------------------------------------------- node inits
__global__ __launch_bounds__(256) void k_init_deg(float* __restrict__ deg, int n) {
  int i = blockIdx.x * blockDim.x + threadIdx.x;
  if (i < n) deg[i] = 1.0f;  // self-loop contributes 1 to every node's degree
}

// ------------------------------------------------------------- degree counts
__global__ __launch_bounds__(256) void k_deg(const int* __restrict__ dst,
                                             float* __restrict__ deg, int e) {
  int i = blockIdx.x * blockDim.x + threadIdx.x;
  if (i < e) atomicAdd(&deg[dst[i]], 1.0f);
}

// --------------------------------------- dinv = rsqrt(deg); s1 self-loop init
__global__ __launch_bounds__(256) void k_dinv_s1(const float* __restrict__ x,
                                                 float* __restrict__ deg_dinv,
                                                 float* __restrict__ s1, int n) {
  int i = blockIdx.x * blockDim.x + threadIdx.x;
  if (i < n) {
    float di = rsqrtf(deg_dinv[i]);
    deg_dinv[i] = di;              // in-place: deg -> dinv
    s1[i] = x[i] * di * di;        // self-loop message (norm = dinv^2)
  }
}

// ------------------- scalar normalized scatter-add: out[d] += v[s]*di[s]*di[d]
__global__ __launch_bounds__(256) void k_scatter(const int* __restrict__ src,
                                                 const int* __restrict__ dst,
                                                 const float* __restrict__ val,
                                                 const float* __restrict__ dinv,
                                                 float* __restrict__ out, int e) {
  int i = blockIdx.x * blockDim.x + threadIdx.x;
  if (i < e) {
    int s = src[i], d = dst[i];
    atomicAdd(&out[d], val[s] * dinv[s] * dinv[d]);
  }
}

// --------------------------------------------------------------------------
// Layer-2 transform via WMMA: for 16 nodes per wave compute
//   m2[i] = sum_k relu(s1[i]*W1[k] + b1[k]) * W2[k]
// A-tile (16x32 f16): A[m][k] = relu(s1[nb+m]*W1[k0+k] + b1[k0+k])
//   per ISA layout: lane (m = lane&15), halves 2v+t map to
//   K = (lane>=16 ? 8:0) + 2v + (v>=4 ? 8:0) + t
// B-tile (32x16 f16): W2 slice, identical for all 16 N-columns:
//   K = (lane>=16 ? 8:0) + 2v + (v>=4 ? 8:0) + t  (same as A)
// Accumulate 16 K-steps; extract column N=0 (lane 0 -> M=0..7, lane 16 -> M=8..15).
// Also seeds s2 with the self-loop term m2[i]*dinv[i]^2.
// --------------------------------------------------------------------------
__global__ __launch_bounds__(256) void k_layer2_wmma(
    const float* __restrict__ s1, const float* __restrict__ dinv,
    const float* __restrict__ W1, const float* __restrict__ b1,
    const float* __restrict__ W2, float* __restrict__ m2,
    float* __restrict__ s2, int n) {
  __shared__ float sW1[HDIM], sb1[HDIM], sW2[HDIM];
  for (int i = threadIdx.x; i < HDIM; i += 256) {
    sW1[i] = W1[i]; sb1[i] = b1[i]; sW2[i] = W2[i];
  }
  __syncthreads();

  const int lane = threadIdx.x & 31;
  const int wave = threadIdx.x >> 5;
  const int nb   = blockIdx.x * 128 + wave * 16;   // 8 waves x 16 nodes
  const int mrow = lane & 15;
  const int hi   = lane >> 4;

  int node_ld = nb + mrow;
  if (node_ld >= n) node_ld = n - 1;               // clamp: keep EXEC all-1s
  const float s1v = s1[node_ld];

  v8f c = {};
  const int kbaseA = hi * 8;
  const int kbaseB = hi * 16;

#pragma unroll
  for (int k0 = 0; k0 < HDIM; k0 += 32) {
    v16h a, b;
#pragma unroll
    for (int v = 0; v < 8; ++v) {
      const int ka = k0 + kbaseA + 2 * v + ((v >= 4) ? 8 : 0);
      float h0 = fmaxf(s1v * sW1[ka]     + sb1[ka],     0.0f);
      float h1 = fmaxf(s1v * sW1[ka + 1] + sb1[ka + 1], 0.0f);
      a[2 * v]     = (_Float16)h0;
      a[2 * v + 1] = (_Float16)h1;
      const int kb = k0 + kbaseA + 2 * v + ((v >= 4) ? 8 : 0);
      b[2 * v]     = (_Float16)sW2[kb];
      b[2 * v + 1] = (_Float16)sW2[kb + 1];
    }
    c = __builtin_amdgcn_wmma_f32_16x16x32_f16(
        /*neg_a=*/false, a, /*neg_b=*/false, b,
        /*c_mod=*/(short)0, c, /*reuse_a=*/false, /*reuse_b=*/false);
  }

  // Extract column N=0 of the 16x16 f32 accumulator.
  if (mrow == 0) {
#pragma unroll
    for (int r = 0; r < 8; ++r) {
      int node = nb + hi * 8 + r;
      if (node < n) {
        float v = c[r];
        m2[node] = v;
        float di = dinv[node];
        s2[node] = v * di * di;   // self-loop contribution of layer 2
      }
    }
  }
}

// ---------------- o = s2 + b2; log_softmax over width-1 axis == o - o (== 0)
__global__ __launch_bounds__(256) void k_out(const float* __restrict__ s2,
                                             const float* __restrict__ b2,
                                             float* __restrict__ out, int n) {
  int i = blockIdx.x * blockDim.x + threadIdx.x;
  if (i < n) {
    float o = s2[i] + b2[0];
    out[i] = o - o;   // exact per IEEE for finite o; NaN-propagating otherwise
  }
}

extern "C" void kernel_launch(void* const* d_in, const int* in_sizes, int n_in,
                              void* d_out, int out_size, void* d_ws, size_t ws_size,
                              hipStream_t stream) {
  const float* x  = (const float*)d_in[0];
  const int*   ei = (const int*)d_in[1];   // [2, E]: row 0 = src, row 1 = dst
  const float* W1 = (const float*)d_in[2];
  const float* b1 = (const float*)d_in[3];
  const float* W2 = (const float*)d_in[4];
  const float* b2 = (const float*)d_in[5];
  float* out = (float*)d_out;

  const int n = in_sizes[0];
  const int e = in_sizes[1] / 2;
  const int* src = ei;
  const int* dst = ei + e;

  float* ws   = (float*)d_ws;
  float* dinv = ws;            // deg, then dinv in place
  float* s1   = ws + (size_t)n;
  float* m2   = ws + (size_t)2 * n;
  float* s2   = ws + (size_t)3 * n;

  const int gn = (n + 255) / 256;
  const int ge = (e + 255) / 256;
  const int gw = (n + 127) / 128;

  k_init_deg   <<<gn, 256, 0, stream>>>(dinv, n);
  k_deg        <<<ge, 256, 0, stream>>>(dst, dinv, e);
  k_dinv_s1    <<<gn, 256, 0, stream>>>(x, dinv, s1, n);
  k_scatter    <<<ge, 256, 0, stream>>>(src, dst, x, dinv, s1, e);
  k_layer2_wmma<<<gw, 256, 0, stream>>>(s1, dinv, W1, b1, W2, m2, s2, n);
  k_scatter    <<<ge, 256, 0, stream>>>(src, dst, m2, dinv, s2, e);
  k_out        <<<gn, 256, 0, stream>>>(s2, b2, out, n);
}
